// CausalSelfAttention_14577119002803
// MI455X (gfx1250) — compile-verified, matrix operand order checked by reading, not run on hardware
//
#include <hip/hip_runtime.h>

#define BB 4
#define TT 2048
#define CC 768
#define HH 12
#define HD 64

typedef _Float16 h16v __attribute__((ext_vector_type(16)));
typedef _Float16 h2v  __attribute__((ext_vector_type(2)));
typedef float    f8v  __attribute__((ext_vector_type(8)));

// WMMA f32 = f16 x f16 + f32, 16x16x32
static __device__ inline f8v wmma_f16(h16v a, h16v b, f8v c) {
  return __builtin_amdgcn_wmma_f32_16x16x32_f16(false, a, false, b, (short)0, c,
                                                false, false);
}

// Load a 16x32 f16 tile in WMMA A-layout (also serves as B when the source is
// [N,K] row-major: pre-transposed weights, K^T from a [key][dim] tile, or
// V^T from a [dim][key] tile).
// Lane l: row = row0 + (l&15); lanes 16-31 handle K offsets +8.
// VGPR j (element pair 2j,2j+1): K = (j>=4?16:0) + (j&3)*2 (+8 for upper half).
static __device__ inline h16v load_tile_rowmajor(const _Float16* base, int ld,
                                                 int row0, int k0, int lane) {
  const _Float16* p =
      base + (size_t)(row0 + (lane & 15)) * ld + k0 + ((lane >> 4) << 3);
  h16v r;
#pragma unroll
  for (int j = 0; j < 8; ++j) {
    int k = ((j & 4) << 2) + ((j & 3) << 1);  // 0,2,4,6,16,18,20,22
    h2v t = *(const h2v*)(p + k);
    r[2 * j] = t[0];
    r[2 * j + 1] = t[1];
  }
  return r;
}

// Issue one stage of async global->LDS copies (ASYNCcnt-tracked, b128):
//   K chunk: 32 keys x 64 dims, contiguous [key][dim] rows -> lk (4 KB)
//   V chunk: 64 dims x 32 keys, strided rows of v^T [dim][T] -> lvT (4 KB)
// 128 threads, 4 async ops per thread per stage.
static __device__ inline void stage_async_kv(const _Float16* gk,
                                             const _Float16* gvT,
                                             _Float16* lk, _Float16* lvT,
                                             int tid) {
  // Low 32 bits of a generic pointer into LDS == AS(3) byte offset (ISA 10.2).
  unsigned lk0 = (unsigned)(size_t)lk;
  unsigned lv0 = (unsigned)(size_t)lvT;
  unsigned long long gk0 = (unsigned long long)(size_t)gk;
  unsigned long long gv0 = (unsigned long long)(size_t)gvT;
#pragma unroll
  for (int i = 0; i < 2; ++i) {
    unsigned idx = (unsigned)(tid + 128 * i);  // 0..255 (16B units)
    unsigned kdst = lk0 + idx * 16u;
    unsigned long long ksrc = gk0 + idx * 16u;
    asm volatile("global_load_async_to_lds_b128 %0, %1, off"
                 :
                 : "v"(kdst), "v"(ksrc)
                 : "memory");
    unsigned d = idx >> 2;        // dim row 0..63
    unsigned part = idx & 3;      // 16B piece within 64B row
    unsigned vdst = lv0 + d * 64u + part * 16u;
    unsigned long long vsrc = gv0 + (unsigned long long)d * (TT * 2u) + part * 16u;
    asm volatile("global_load_async_to_lds_b128 %0, %1, off"
                 :
                 : "v"(vdst), "v"(vsrc)
                 : "memory");
  }
}

__global__ void cvt_f32_to_f16(const float* __restrict__ in,
                               _Float16* __restrict__ out, int n) {
  int i = blockIdx.x * blockDim.x + threadIdx.x;
  if (i < n) out[i] = (_Float16)in[i];
}

// out[n*K + k] = (f16) in[k*N + n]   (transpose [K,N] -> [N,K] + convert)
__global__ void transpose_cvt(const float* __restrict__ in,
                              _Float16* __restrict__ out, int K, int N) {
  int i = blockIdx.x * blockDim.x + threadIdx.x;
  if (i >= N * K) return;
  int n = i / K;
  int k = i - n * K;
  out[i] = (_Float16)in[(size_t)k * N + n];
}

// QKV projection: [8192,768] x Wt[2304,768]^T + bias.
// q,k scattered to [B,H,T,64] f16; v scattered TRANSPOSED to [B,H,64,T] f16
// so the attention kernel can read V^T fragments with contiguous pairs.
// 1 wave = 32 rows x 64 cols (2 M-tiles x 4 N-tiles, B fragment reused twice).
__global__ void qkv_gemm(const _Float16* __restrict__ x16,
                         const _Float16* __restrict__ wtA,
                         const float* __restrict__ bias,
                         _Float16* __restrict__ q16, _Float16* __restrict__ k16,
                         _Float16* __restrict__ v16t) {
  int lane = threadIdx.x & 31;
  int wave = threadIdx.x >> 5;
  int row0 = (blockIdx.x * 4 + wave) * 32;
  int col0 = blockIdx.y * 64;
  f8v acc[2][4] = {};
  for (int k0 = 0; k0 < CC; k0 += 32) {
    h16v a0 = load_tile_rowmajor(x16, CC, row0, k0, lane);
    h16v a1 = load_tile_rowmajor(x16, CC, row0 + 16, k0, lane);
#pragma unroll
    for (int nt = 0; nt < 4; ++nt) {
      h16v b = load_tile_rowmajor(wtA, CC, col0 + nt * 16, k0, lane);
      acc[0][nt] = wmma_f16(a0, b, acc[0][nt]);
      acc[1][nt] = wmma_f16(a1, b, acc[1][nt]);
    }
  }
  int halfsel = (lane >> 4) << 3;
#pragma unroll
  for (int nt = 0; nt < 4; ++nt) {
    int colN = col0 + nt * 16 + (lane & 15);
    float bv = bias[colN];
    int which = colN / CC;          // 0=q, 1=k, 2=v (uniform per wave)
    int rem = colN - which * CC;
    int head = rem >> 6;            // uniform per wave
    int d = rem & 63;
    if (which == 2) {
#pragma unroll
      for (int mt = 0; mt < 2; ++mt) {
#pragma unroll
        for (int r = 0; r < 8; ++r) {
          int row = row0 + mt * 16 + r + halfsel;
          int bi = row >> 11;
          int t = row & (TT - 1);
          v16t[(((size_t)(bi * HH + head)) * HD + d) * TT + t] =
              (_Float16)(acc[mt][nt][r] + bv);
        }
      }
    } else {
      _Float16* dst = which == 0 ? q16 : k16;
#pragma unroll
      for (int mt = 0; mt < 2; ++mt) {
#pragma unroll
        for (int r = 0; r < 8; ++r) {
          int row = row0 + mt * 16 + r + halfsel;
          int bi = row >> 11;
          int t = row & (TT - 1);
          dst[(((size_t)(bi * HH + head)) * TT + t) * HD + d] =
              (_Float16)(acc[mt][nt][r] + bv);
        }
      }
    }
  }
}

// Causal flash attention. Block = 4 waves, each owning one 16-query tile;
// K ([key][dim]) and V^T ([dim][key]) 32-key chunks are async-staged into
// double-buffered LDS shared by all four waves. Output merged heads, f16.
__global__ void attn_kernel(const _Float16* __restrict__ q16,
                            const _Float16* __restrict__ k16,
                            const _Float16* __restrict__ v16t,
                            _Float16* __restrict__ y16) {
  __shared__ _Float16 kbuf[2][32 * HD];   // 2 x 4 KB, [key][dim]
  __shared__ _Float16 vbuf[2][HD * 32];   // 2 x 4 KB, [dim][key]
  __shared__ _Float16 pld[4][16 * 32];    // per-wave P staging, 4 x 1 KB
  int tid = threadIdx.x;
  int lane = tid & 31;
  int wave = tid >> 5;
  int q0 = (blockIdx.x * 4 + wave) * 16;
  int bh = blockIdx.y;
  int head = bh % HH;
  int bi = bh / HH;
  const _Float16* qbase = q16 + (size_t)bh * TT * HD;
  const _Float16* kbase = k16 + (size_t)bh * TT * HD;
  const _Float16* vtbase = v16t + (size_t)bh * HD * TT;
  _Float16* mypld = &pld[wave][0];

  h16v qa0 = load_tile_rowmajor(qbase, HD, q0, 0, lane);
  h16v qa1 = load_tile_rowmajor(qbase, HD, q0, 32, lane);

  f8v acc[4] = {};
  float m[8], lsum[8];
  int halfsel = (lane >> 4) << 3;
#pragma unroll
  for (int r = 0; r < 8; ++r) {
    m[r] = -__builtin_inff();
    lsum[r] = 0.f;
  }

  int kend = q0 + 16;                      // this wave's causal bound
  int kend_blk = blockIdx.x * 64 + 64;     // last wave's bound (uniform)

  stage_async_kv(kbase, vtbase, kbuf[0], vbuf[0], tid);

  for (int kc = 0; kc < kend_blk; kc += 32) {
    int cur = (kc >> 5) & 1;
    if (kc + 32 < kend_blk) {
      stage_async_kv(kbase + (size_t)(kc + 32) * HD, vtbase + (kc + 32),
                     kbuf[cur ^ 1], vbuf[cur ^ 1], tid);
      asm volatile("s_wait_asynccnt 0x4" ::: "memory");
    } else {
      asm volatile("s_wait_asynccnt 0x0" ::: "memory");
    }
    __syncthreads();

    if (kc < kend) {  // wave-uniform: EXEC stays all-ones for WMMA
      const _Float16* kld = kbuf[cur];
      const _Float16* vld = vbuf[cur];
      // S = Q K^T for 16 queries x 32 keys (two 16-col tiles, K-dim = 64).
      f8v s0 = {}, s1 = {};
      {
        h16v b00 = load_tile_rowmajor(kld, HD, 0, 0, lane);
        h16v b01 = load_tile_rowmajor(kld, HD, 0, 32, lane);
        s0 = wmma_f16(qa0, b00, s0);
        s0 = wmma_f16(qa1, b01, s0);
        h16v b10 = load_tile_rowmajor(kld, HD, 16, 0, lane);
        h16v b11 = load_tile_rowmajor(kld, HD, 16, 32, lane);
        s1 = wmma_f16(qa0, b10, s1);
        s1 = wmma_f16(qa1, b11, s1);
      }

      int key0 = kc + (lane & 15);
      int key1 = key0 + 16;
      float p0[8], p1[8];
#pragma unroll
      for (int r = 0; r < 8; ++r) {
        int qrow = q0 + r + halfsel;
        float v0 = s0[r] * 0.125f;  // 1/sqrt(64)
        float v1 = s1[r] * 0.125f;
        if (key0 > qrow) v0 = -__builtin_inff();
        if (key1 > qrow) v1 = -__builtin_inff();
        float rm = fmaxf(v0, v1);
#pragma unroll
        for (int msk = 1; msk < 16; msk <<= 1)
          rm = fmaxf(rm, __shfl_xor(rm, msk, 32));
        float mn = fmaxf(m[r], rm);
        float alpha = __expf(m[r] - mn);
        float e0 = __expf(v0 - mn);
        float e1 = __expf(v1 - mn);
        float rs = e0 + e1;
#pragma unroll
        for (int msk = 1; msk < 16; msk <<= 1) rs += __shfl_xor(rs, msk, 32);
        lsum[r] = lsum[r] * alpha + rs;
        m[r] = mn;
#pragma unroll
        for (int nt = 0; nt < 4; ++nt) acc[nt][r] *= alpha;
        p0[r] = e0;
        p1[r] = e1;
      }

      // Reshape P (C/D layout -> A layout) through this wave's LDS slice.
      // LDS ops are in-order within a wave; wave_barrier stops reordering.
#pragma unroll
      for (int r = 0; r < 8; ++r) {
        int rowM = r + halfsel;
        mypld[rowM * 32 + (lane & 15)] = (_Float16)p0[r];
        mypld[rowM * 32 + 16 + (lane & 15)] = (_Float16)p1[r];
      }
      __builtin_amdgcn_wave_barrier();
      h16v pa = load_tile_rowmajor(mypld, 32, 0, 0, lane);
#pragma unroll
      for (int nt = 0; nt < 4; ++nt) {
        // V^T tile is [dim][key]: rows = output dims, K = keys -> same
        // contiguous-pair fragment loader as everything else.
        h16v bv = load_tile_rowmajor(vld, 32, nt * 16, 0, lane);
        acc[nt] = wmma_f16(pa, bv, acc[nt]);
      }
    }
    __syncthreads();
  }

  float inv[8];
#pragma unroll
  for (int r = 0; r < 8; ++r) inv[r] = 1.f / lsum[r];
#pragma unroll
  for (int nt = 0; nt < 4; ++nt) {
    int col = head * HD + nt * 16 + (lane & 15);
#pragma unroll
    for (int r = 0; r < 8; ++r) {
      int t = q0 + r + halfsel;
      y16[((size_t)(bi * TT + t)) * CC + col] = (_Float16)(acc[nt][r] * inv[r]);
    }
  }
}

// Output projection: [8192,768] x WtP[768,768]^T + bias -> f32 out.
// Same 32x64-per-wave register blocking as qkv_gemm.
__global__ void proj_gemm(const _Float16* __restrict__ y16,
                          const _Float16* __restrict__ wtP,
                          const float* __restrict__ bias,
                          float* __restrict__ out) {
  int lane = threadIdx.x & 31;
  int wave = threadIdx.x >> 5;
  int row0 = (blockIdx.x * 4 + wave) * 32;
  int col0 = blockIdx.y * 64;
  f8v acc[2][4] = {};
  for (int k0 = 0; k0 < CC; k0 += 32) {
    h16v a0 = load_tile_rowmajor(y16, CC, row0, k0, lane);
    h16v a1 = load_tile_rowmajor(y16, CC, row0 + 16, k0, lane);
#pragma unroll
    for (int nt = 0; nt < 4; ++nt) {
      h16v b = load_tile_rowmajor(wtP, CC, col0 + nt * 16, k0, lane);
      acc[0][nt] = wmma_f16(a0, b, acc[0][nt]);
      acc[1][nt] = wmma_f16(a1, b, acc[1][nt]);
    }
  }
  int halfsel = (lane >> 4) << 3;
#pragma unroll
  for (int nt = 0; nt < 4; ++nt) {
    int colN = col0 + nt * 16 + (lane & 15);
    float bv = bias[colN];
#pragma unroll
    for (int mt = 0; mt < 2; ++mt) {
#pragma unroll
      for (int r = 0; r < 8; ++r) {
        int row = row0 + mt * 16 + r + halfsel;
        out[(size_t)row * CC + colN] = acc[mt][nt][r] + bv;
      }
    }
  }
}

extern "C" void kernel_launch(void* const* d_in, const int* in_sizes, int n_in,
                              void* d_out, int out_size, void* d_ws,
                              size_t ws_size, hipStream_t stream) {
  const float* x = (const float*)d_in[0];
  const float* W_attn = (const float*)d_in[1];
  const float* b_attn = (const float*)d_in[2];
  const float* W_proj = (const float*)d_in[3];
  const float* b_proj = (const float*)d_in[4];
  float* out = (float*)d_out;

  // Workspace carve-up (f16 elements), ~68 MB total.
  _Float16* x16 = (_Float16*)d_ws;                      // [8192,768]
  _Float16* wtA = x16 + (size_t)BB * TT * CC;           // [2304,768]
  _Float16* wtP = wtA + (size_t)3 * CC * CC;            // [768,768]
  _Float16* q16 = wtP + (size_t)CC * CC;                // [B,H,T,64]
  _Float16* k16 = q16 + (size_t)BB * HH * TT * HD;      // [B,H,T,64]
  _Float16* v16t = k16 + (size_t)BB * HH * TT * HD;     // [B,H,64,T]
  _Float16* y16 = v16t + (size_t)BB * HH * TT * HD;     // [8192,768]

  int nx = BB * TT * CC;
  cvt_f32_to_f16<<<(nx + 255) / 256, 256, 0, stream>>>(x, x16, nx);
  int nA = CC * 3 * CC;
  transpose_cvt<<<(nA + 255) / 256, 256, 0, stream>>>(W_attn, wtA, CC, 3 * CC);
  int nP = CC * CC;
  transpose_cvt<<<(nP + 255) / 256, 256, 0, stream>>>(W_proj, wtP, CC, CC);

  qkv_gemm<<<dim3(BB * TT / 128, 3 * CC / 64), 128, 0, stream>>>(
      x16, wtA, b_attn, q16, k16, v16t);
  attn_kernel<<<dim3(TT / 64, BB * HH), 128, 0, stream>>>(q16, k16, v16t, y16);
  proj_gemm<<<dim3(BB * TT / 128, CC / 64), 128, 0, stream>>>(y16, wtP, b_proj,
                                                              out);
}
